// DGCN_37271726195346
// MI455X (gfx1250) — matrix operand order repaired and checked by reading, not compiled, not run on hardware
//
#include <hip/hip_runtime.h>
#include <hip/hip_bf16.h>

// ---------------------------------------------------------------------------
// DGCN pipeline for MI455X (gfx1250, wave32, WMMA bf16 16x16x32 w/ f32 acc)
//
// dims: B=1, C=48 (pad K to 64), H=W=160 (N=25600 pixels), K=8 neighbors,
//       WS=32 (25 windows), RANK=12, COUT=48, M=3, DELTA=10, LEAK=0.01
// ---------------------------------------------------------------------------

typedef __attribute__((ext_vector_type(16))) __bf16 v16bf;
typedef __attribute__((ext_vector_type(8)))  float  v8f;

#define NPIX   25600
#define NC     48
#define KPAD   64
#define NWIN   5          // windows per side (160/32)
#define TOPK   8
#define NEXP   576        // RANK*COUT = RANK*C

__device__ __forceinline__ unsigned short f2bf(float x) {
  unsigned u = __float_as_uint(x);
  unsigned r = u + 0x7fffu + ((u >> 16) & 1u);   // round-to-nearest-even
  return (unsigned short)(r >> 16);
}

__device__ __forceinline__ v8f z8() {
  v8f z = {0.f, 0.f, 0.f, 0.f, 0.f, 0.f, 0.f, 0.f};
  return z;
}

// A operand (16x32 bf16, M = lane&15): lanes 0-15 hold K {kb..kb+7, kb+16..kb+23},
// lanes 16-31 hold K {kb+8..kb+15, kb+24..kb+31}.  row64 = 64-entry bf16 row.
__device__ __forceinline__ v16bf frag_a(const unsigned short* row64, int kbase, int lane) {
  union { v16bf v; unsigned short s[16]; uint4 q[2]; } u;
  int ks = kbase + ((lane >> 4) & 1) * 8;
  u.q[0] = *(const uint4*)(row64 + ks);
  u.q[1] = *(const uint4*)(row64 + ks + 16);
  return u.v;
}

// B operand (32x16 bf16, N = lane&15): same K split as A (lanes 0-15 K {kb..kb+7, kb+16..kb+23}, lanes 16-31 K {kb+8..kb+15, kb+24..kb+31}).
// col64 = 64-entry bf16 "B-transposed" row for column N (i.e. stored [N][K]).
__device__ __forceinline__ v16bf frag_b(const unsigned short* col64, int kbase, int lane) {
  union { v16bf v; unsigned short s[16]; uint4 q[2]; } u;
  int ks = kbase + ((lane >> 4) & 1) * 8;
  u.q[0] = *(const uint4*)(col64 + ks);
  u.q[1] = *(const uint4*)(col64 + ks + 16);
  return u.v;
}

__device__ __forceinline__ v8f wmma_bf16(v16bf a, v16bf b, v8f c) {
  // (neg_a, A, neg_b, B, c_mod, C, reuse_a, reuse_b)
  return __builtin_amdgcn_wmma_f32_16x16x32_bf16(false, a, false, b, (short)0, c,
                                                 false, false);
}

// ---------------------------------------------------------------------------
// Kernel 1: pack h (f32 [C][N]) -> bf16 [pixel][64] (zero K-pad) + sumsq.
// ---------------------------------------------------------------------------
__global__ __launch_bounds__(256) void prep_h_kernel(const float* __restrict__ h,
                                                     unsigned short* __restrict__ hbf,
                                                     float* __restrict__ sqg) {
  int p = blockIdx.x * 256 + threadIdx.x;
  if (p >= NPIX) return;
  float s = 0.f;
  #pragma unroll 4
  for (int c = 0; c < NC; ++c) {
    float v = h[c * NPIX + p];
    s += v * v;
    hbf[p * KPAD + c] = f2bf(v);
  }
  #pragma unroll
  for (int c = NC; c < KPAD; ++c) hbf[p * KPAD + c] = 0;
  sqg[p] = s;
}

// ---------------------------------------------------------------------------
// Kernel 2: expand weights into bf16 B-transposed [N][64] layouts.
//   W0b : 48x64   from W0 (48x48)
//   WLb : 576x64  WLbig[e][q] = WL[e/3][0][(q-e%3) mod 48]   (circ_dense)
//   WRb : 576x64  same from WR
//   Wkb : 16x64   rows 0..11 = Wk, rest zero
// ---------------------------------------------------------------------------
__global__ __launch_bounds__(256) void prep_w_kernel(const float* __restrict__ W0,
                                                     const float* __restrict__ WL,
                                                     const float* __restrict__ WR,
                                                     const float* __restrict__ Wk,
                                                     unsigned short* __restrict__ W0b,
                                                     unsigned short* __restrict__ WLb,
                                                     unsigned short* __restrict__ WRb,
                                                     unsigned short* __restrict__ Wkb) {
  int idx = blockIdx.x * 256 + threadIdx.x;
  if (idx >= 1216 * KPAD) return;
  int row = idx >> 6, q = idx & 63;
  if (row < 48) {
    W0b[row * KPAD + q] = (q < NC) ? f2bf(W0[row * NC + q]) : (unsigned short)0;
  } else if (row < 48 + NEXP) {
    int e = row - 48, i = e / 3, j = e - i * 3;
    WLb[e * KPAD + q] = (q < NC) ? f2bf(WL[i * NC + ((q - j + NC) % NC)]) : (unsigned short)0;
  } else if (row < 48 + 2 * NEXP) {
    int e = row - (48 + NEXP), i = e / 3, j = e - i * 3;
    WRb[e * KPAD + q] = (q < NC) ? f2bf(WR[i * NC + ((q - j + NC) % NC)]) : (unsigned short)0;
  } else {
    int e = row - (48 + 2 * NEXP);
    Wkb[e * KPAD + q] = (e < 12 && q < NC) ? f2bf(Wk[e * NC + q]) : (unsigned short)0;
  }
}

// ---------------------------------------------------------------------------
// Kernel 3: per-window top-K.  grid = 25 windows * 4 chunks, 512 thr (16 waves).
// The whole 1024x64 bf16 window (128 KB, fits gfx1250's 320 KB WGP LDS) is
// staged once per block with GLOBAL_LOAD_ASYNC_TO_LDS_B128 (ASYNCcnt path),
// then every wave feeds its WMMA A/B fragments from LDS.
// Wave owns one 16-row tile; 64 column tiles, each a 16x16 WMMA Gram tile.
// d = sq_i + sq_j - 2*G, 3x3 local mask, register insertion top-8 per row.
// ---------------------------------------------------------------------------
__global__ __launch_bounds__(512) void topk_kernel(const unsigned short* __restrict__ hbf,
                                                   const float* __restrict__ sqg,
                                                   int* __restrict__ edge) {
  __shared__ __attribute__((aligned(16))) unsigned short Xs[1024][KPAD];  // 128 KB
  __shared__ __attribute__((aligned(16))) float sqW[1024];                //   4 KB
  __shared__ __attribute__((aligned(16))) float dscr[16][256];            //  16 KB

  const int win = blockIdx.x >> 2, chunk = blockIdx.x & 3;
  const int wi = win / NWIN, wj = win % NWIN;
  const int py0 = wi * 32, px0 = wj * 32;
  const int tid = threadIdx.x, lane = tid & 31, wave = tid >> 5;

  // Async-stage window features into LDS: 1024 rows x 128 B, 16 B per issue.
  for (int t = tid; t < 1024 * 8; t += 512) {
    int row = t >> 3, ch = (t & 7) * 8;            // 8 bf16 = 16 B chunk
    int p = (py0 + (row >> 5)) * 160 + px0 + (row & 31);
    const unsigned short* g = hbf + (size_t)p * KPAD + ch;
    unsigned lds = (unsigned)(size_t)(&Xs[row][ch]);
    asm volatile("global_load_async_to_lds_b128 %0, %1, off"
                 :: "v"(lds), "v"(g)
                 : "memory");
  }
  for (int n = tid; n < 1024; n += 512) {
    int p = (py0 + (n >> 5)) * 160 + px0 + (n & 31);
    sqW[n] = sqg[p];
  }
  asm volatile("s_wait_asynccnt 0x0" ::: "memory"); // drain this wave's async copies
  __syncthreads();                                  // all waves' copies visible

  const int rt = chunk * 16 + wave;                 // row tile 0..63
  const int colh = lane & 15, rbase = (lane >> 4) * 8;

  v16bf a0 = frag_a(&Xs[rt * 16 + colh][0], 0, lane);
  v16bf a1 = frag_a(&Xs[rt * 16 + colh][0], 32, lane);

  float bd[TOPK]; int bi[TOPK];
  #pragma unroll
  for (int j = 0; j < TOPK; ++j) { bd[j] = 3.0e38f; bi[j] = 0; }
  float* dt = dscr[wave];

  for (int ct = 0; ct < 64; ++ct) {
    v16bf b0 = frag_b(&Xs[ct * 16 + colh][0], 0, lane);
    v16bf b1 = frag_b(&Xs[ct * 16 + colh][0], 32, lane);
    v8f c = z8();
    c = wmma_bf16(a0, b0, c);
    c = wmma_bf16(a1, b1, c);
    #pragma unroll
    for (int v = 0; v < 8; ++v) {
      int m = rt * 16 + rbase + v;
      int n = ct * 16 + colh;
      float d = sqW[m] + sqW[n] - 2.0f * c[v];
      int dr = (m >> 5) - (n >> 5); dr = dr < 0 ? -dr : dr;
      int dc = (m & 31) - (n & 31); dc = dc < 0 ? -dc : dc;
      if (dr <= 1 && dc <= 1) d = 1.0e10f;          // localMask (ks=3), incl. self
      dt[(rbase + v) * 16 + colh] = d;
    }
    __builtin_amdgcn_wave_barrier();
    if (lane < 16) {
      #pragma unroll
      for (int j = 0; j < 16; ++j) {
        float d = dt[lane * 16 + j];
        int n = ct * 16 + j;
        if (d < bd[TOPK - 1]) {
          bd[TOPK - 1] = d; bi[TOPK - 1] = n;
          #pragma unroll
          for (int t = TOPK - 1; t > 0; --t) {
            if (bd[t] < bd[t - 1]) {
              float td = bd[t]; bd[t] = bd[t - 1]; bd[t - 1] = td;
              int ti = bi[t]; bi[t] = bi[t - 1]; bi[t - 1] = ti;
            }
          }
        }
      }
    }
    __builtin_amdgcn_wave_barrier();
  }

  if (lane < 16) {
    int m = rt * 16 + lane;
    int p = (py0 + (m >> 5)) * 160 + px0 + (m & 31);
    #pragma unroll
    for (int k = 0; k < TOPK; ++k) {
      int n = bi[k];
      edge[k * NPIX + p] = (py0 + (n >> 5)) * 160 + px0 + (n & 31);
    }
  }
}

// ---------------------------------------------------------------------------
// Kernel 4: fused low-rank ECC.  128 thr = 4 waves; each wave owns 2 pixels x
// 8 neighbors = 16 edge rows (row = half*8 + k).  Per wave:
//   label/vertex gather -> theta GEMM (leaky relu) -> kap GEMM ->
//   tR GEMM fused with s[r]=sum_c tR[c][r]*v[c]  (LDS atomics) ->
//   tmp = kap*s -> tL GEMM fused with out[o]=sum_r tL[o][r]*tmp[r] ->
//   out[pixel][o] = sum_k gamma_k * out_k   (plain store, no global atomics)
// ---------------------------------------------------------------------------
#define CW 4
__global__ __launch_bounds__(128) void ecc_kernel(const float* __restrict__ h,
                                                  const int* __restrict__ edge,
                                                  const unsigned short* __restrict__ W0b,
                                                  const float* __restrict__ b0v,
                                                  const unsigned short* __restrict__ WLb,
                                                  const unsigned short* __restrict__ WRb,
                                                  const unsigned short* __restrict__ Wkb,
                                                  const float* __restrict__ bkv,
                                                  float* __restrict__ outAcc) {
  __shared__ __attribute__((aligned(16))) unsigned short labA[CW][16][KPAD];
  __shared__ __attribute__((aligned(16))) unsigned short thA[CW][16][KPAD];
  __shared__ __attribute__((aligned(16))) float vert[CW][16][NC];
  __shared__ __attribute__((aligned(16))) float sacc[CW][16][16];
  __shared__ __attribute__((aligned(16))) float oacc[CW][16][NC];
  __shared__ __attribute__((aligned(16))) float kapS[CW][16][12];
  __shared__ __attribute__((aligned(16))) float gamS[CW][16];

  const int tid = threadIdx.x, lane = tid & 31, w = tid >> 5;
  const int p0 = (blockIdx.x * CW + w) * 2;         // first of 2 pixels
  const int colh = lane & 15, rbase = (lane >> 4) * 8;

  { // zero accumulators + K-pad regions
    float* pz = &sacc[w][0][0];
    for (int i = lane; i < 256; i += 32) pz[i] = 0.f;
    pz = &oacc[w][0][0];
    for (int i = lane; i < 16 * NC; i += 32) pz[i] = 0.f;
    if (lane < 16) gamS[w][lane] = 0.f;
    for (int i = lane; i < 256; i += 32) {          // pad cols 48..63
      int row = i >> 4, c = NC + (i & 15);
      labA[w][row][c] = 0; thA[w][row][c] = 0;
    }
  }
  __builtin_amdgcn_wave_barrier();

  // gather labels / vertices; accumulate ||label||^2
  for (int row = 0; row < 16; ++row) {
    int pix = p0 + (row >> 3);
    int k = row & 7;
    int nb = edge[k * NPIX + pix];
    float part = 0.f;
    for (int c = lane; c < NC; c += 32) {
      float vv = h[c * NPIX + nb];
      float cv = h[c * NPIX + pix];
      vert[w][row][c] = vv;
      float lb = vv - cv;
      labA[w][row][c] = f2bf(lb);
      part += lb * lb;
    }
    atomicAdd(&gamS[w][row], part);
  }
  __builtin_amdgcn_wave_barrier();
  if (lane < 16) gamS[w][lane] = expf(-gamS[w][lane] * 0.1f);   // gamma, DELTA=10
  __builtin_amdgcn_wave_barrier();

  // theta = leaky_relu(label @ W0^T + b0)
  {
    v16bf a0 = frag_a(&labA[w][colh][0], 0, lane);
    v16bf a1 = frag_a(&labA[w][colh][0], 32, lane);
    #pragma unroll
    for (int nt = 0; nt < 3; ++nt) {
      const unsigned short* bp = W0b + (size_t)(nt * 16 + colh) * KPAD;
      v8f c = z8();
      c = wmma_bf16(a0, frag_b(bp, 0, lane), c);
      c = wmma_bf16(a1, frag_b(bp, 32, lane), c);
      int col = nt * 16 + colh;
      float bias0 = b0v[col];
      #pragma unroll
      for (int v = 0; v < 8; ++v) {
        float t = c[v] + bias0;
        t = t > 0.f ? t : 0.01f * t;                // LEAK
        thA[w][rbase + v][col] = f2bf(t);
      }
    }
  }
  __builtin_amdgcn_wave_barrier();

  v16bf ta0 = frag_a(&thA[w][colh][0], 0, lane);
  v16bf ta1 = frag_a(&thA[w][colh][0], 32, lane);

  // kap = theta @ Wk^T + bk   (N padded 12 -> 16)
  {
    const unsigned short* bp = Wkb + (size_t)colh * KPAD;
    v8f c = z8();
    c = wmma_bf16(ta0, frag_b(bp, 0, lane), c);
    c = wmma_bf16(ta1, frag_b(bp, 32, lane), c);
    if (colh < 12) {
      float bk_ = bkv[colh];
      #pragma unroll
      for (int v = 0; v < 8; ++v) kapS[w][rbase + v][colh] = c[v] + bk_;
    }
  }
  __builtin_amdgcn_wave_barrier();

  // tR GEMM fused with s[r] += tR[c][r] * vertex[c]   (col n = c*12 + r)
  for (int nt = 0; nt < 36; ++nt) {
    const unsigned short* bp = WRb + (size_t)(nt * 16 + colh) * KPAD;
    v8f c = z8();
    c = wmma_bf16(ta0, frag_b(bp, 0, lane), c);
    c = wmma_bf16(ta1, frag_b(bp, 32, lane), c);
    int n = nt * 16 + colh;
    int cch = n / 12, r = n - cch * 12;
    #pragma unroll
    for (int v = 0; v < 8; ++v)
      atomicAdd(&sacc[w][rbase + v][r], c[v] * vert[w][rbase + v][cch]);
  }
  __builtin_amdgcn_wave_barrier();
  if (lane < 16) {                                  // tmp = kap * s (in place)
    #pragma unroll
    for (int r = 0; r < 12; ++r) sacc[w][lane][r] *= kapS[w][lane][r];
  }
  __builtin_amdgcn_wave_barrier();

  // tL GEMM fused with out[o] += tL[o][r] * tmp[r]   (col n = o*12 + r)
  for (int nt = 0; nt < 36; ++nt) {
    const unsigned short* bp = WLb + (size_t)(nt * 16 + colh) * KPAD;
    v8f c = z8();
    c = wmma_bf16(ta0, frag_b(bp, 0, lane), c);
    c = wmma_bf16(ta1, frag_b(bp, 32, lane), c);
    int n = nt * 16 + colh;
    int o = n / 12, r = n - o * 12;
    #pragma unroll
    for (int v = 0; v < 8; ++v)
      atomicAdd(&oacc[w][rbase + v][o], c[v] * sacc[w][rbase + v][r]);
  }
  __builtin_amdgcn_wave_barrier();

  // out[pixel][o] = sum_k gamma_k * out_k  (k-sum wave-local -> plain stores)
  #pragma unroll
  for (int half = 0; half < 2; ++half) {
    int pix = p0 + half;
    for (int o = lane; o < NC; o += 32) {
      float s = 0.f;
      #pragma unroll
      for (int kk = 0; kk < 8; ++kk) {
        int row = half * 8 + kk;
        s += gamS[w][row] * oacc[w][row][o];
      }
      outAcc[o * NPIX + pix] = s;
    }
  }
}

// ---------------------------------------------------------------------------
// Kernel 5: out = (ecc_sum/8 + reflect-pad 3x3 conv)/2 + bias   (in place)
// ---------------------------------------------------------------------------
__global__ __launch_bounds__(256) void combine_kernel(const float* __restrict__ h,
                                                      const float* __restrict__ Wconv,
                                                      const float* __restrict__ biasv,
                                                      float* __restrict__ out) {
  int idx = blockIdx.x * 256 + threadIdx.x;
  if (idx >= NC * NPIX) return;
  int o = idx / NPIX;
  int p = idx - o * NPIX;
  int y = p / 160, x = p - y * 160;
  float acc = 0.f;
  #pragma unroll 1
  for (int c = 0; c < NC; ++c) {
    const float* hc = h + c * NPIX;
    const float* wc = Wconv + (size_t)(o * NC + c) * 9;
    #pragma unroll
    for (int dy = 0; dy < 3; ++dy) {
      int yy = y + dy - 1; yy = yy < 0 ? -yy : (yy > 159 ? 318 - yy : yy);
      #pragma unroll
      for (int dx = 0; dx < 3; ++dx) {
        int xx = x + dx - 1; xx = xx < 0 ? -xx : (xx > 159 ? 318 - xx : xx);
        acc += hc[yy * 160 + xx] * wc[dy * 3 + dx];
      }
    }
  }
  float s = out[idx];                                // ecc sum over 8 neighbors
  out[idx] = (s * 0.125f + acc) * 0.5f + biasv[o];
}

// ---------------------------------------------------------------------------
extern "C" void kernel_launch(void* const* d_in, const int* in_sizes, int n_in,
                              void* d_out, int out_size, void* d_ws, size_t ws_size,
                              hipStream_t stream) {
  (void)in_sizes; (void)n_in; (void)out_size; (void)ws_size;
  const float* h     = (const float*)d_in[0];
  const float* W0    = (const float*)d_in[1];
  const float* b0v   = (const float*)d_in[2];
  const float* WL    = (const float*)d_in[3];
  const float* WR    = (const float*)d_in[4];
  const float* Wk    = (const float*)d_in[5];
  const float* bkv   = (const float*)d_in[6];
  const float* Wconv = (const float*)d_in[7];
  const float* biasv = (const float*)d_in[8];
  float* out = (float*)d_out;

  // workspace carve-out (~4.4 MB total, 256B aligned slices)
  char* wp = (char*)d_ws;
  auto carve = [&](size_t bytes) {
    char* r = wp;
    wp += (bytes + 255) & ~(size_t)255;
    return r;
  };
  unsigned short* hbf = (unsigned short*)carve((size_t)NPIX * KPAD * 2);
  float*          sqg = (float*)carve((size_t)NPIX * 4);
  unsigned short* W0b = (unsigned short*)carve(48 * KPAD * 2);
  unsigned short* WLb = (unsigned short*)carve((size_t)NEXP * KPAD * 2);
  unsigned short* WRb = (unsigned short*)carve((size_t)NEXP * KPAD * 2);
  unsigned short* Wkb = (unsigned short*)carve(16 * KPAD * 2);
  int*            edg = (int*)carve((size_t)TOPK * NPIX * 4);

  prep_h_kernel<<<NPIX / 256, 256, 0, stream>>>(h, hbf, sqg);
  prep_w_kernel<<<(1216 * KPAD) / 256, 256, 0, stream>>>(W0, WL, WR, Wk, W0b, WLb, WRb, Wkb);
  topk_kernel<<<25 * 4, 512, 0, stream>>>(hbf, sqg, edg);
  ecc_kernel<<<NPIX / (CW * 2), 128, 0, stream>>>(h, edg, W0b, b0v, WLb, WRb, Wkb, bkv, out);
  combine_kernel<<<(NC * NPIX) / 256, 256, 0, stream>>>(h, Wconv, biasv, out);
}
